// TemperatureAwareMTLGNN_52905407152890
// MI455X (gfx1250) — matrix operand order repaired and checked by reading, not compiled, not run on hardware
//
#include <hip/hip_runtime.h>
#include <hip/hip_bf16.h>

// ---------------- problem constants ----------------
#define N_NODES 131072
#define E_EDGES 1048576
#define G_GRAPHS 4096
#define IN_F 39
#define H_F 96
#define OUT_F 192
#define ED_F 10
#define TED_F 64
#define TASKS_F 6
#define TSTEPS 2

// ---------------- WMMA types / helpers ----------------
typedef __attribute__((ext_vector_type(16))) __bf16 bf16x16;
typedef __attribute__((ext_vector_type(8)))  float  f32x8;

union ABu {
  unsigned short u[16];
  uint4 q[2];
  bf16x16 v;
};

__device__ __forceinline__ unsigned short f2bf(float f) {
  unsigned u = __float_as_uint(f);
  u += 0x7FFFu + ((u >> 16) & 1u);           // round-to-nearest-even
  return (unsigned short)(u >> 16);
}
__device__ __forceinline__ float bf2f(unsigned short h) {
  return __uint_as_float(((unsigned)h) << 16);
}
__device__ __forceinline__ float leakyf(float x) { return x > 0.f ? x : 0.01f * x; }
__device__ __forceinline__ float eluf(float x)   { return x > 0.f ? x : (expf(x) - 1.f); }
__device__ __forceinline__ float sigm(float x)   { return 1.f / (1.f + expf(-x)); }

__device__ __forceinline__ void pack4(unsigned short* d, float4 v) {
  ushort4 p;
  p.x = f2bf(v.x); p.y = f2bf(v.y); p.z = f2bf(v.z); p.w = f2bf(v.w);
  *(ushort4*)d = p;
}

// A fragment: lane(m=lane&15, hi) holds K runs [k0+hi*8 .. +7] and [k0+16+hi*8 .. +7]
__device__ __forceinline__ void load_a_frag(ABu& a, const unsigned short* base,
                                            int stride, int row, int k0, int hi) {
  const unsigned short* p = base + (size_t)row * stride + k0 + hi * 8;
  a.q[0] = *(const uint4*)p;
  a.q[1] = *(const uint4*)(p + 16);
}
// B fragment: lane(n=lane&15, hi) holds K runs [k0+hi*8 .. +7] and [k0+16+hi*8 .. +7]
__device__ __forceinline__ void load_b_frag(ABu& b, const unsigned short* base,
                                            int stride, int col, int k0, int hi) {
  const unsigned short* p = base + (size_t)col * stride + k0 + hi * 8;
  b.q[0] = *(const uint4*)p;
  b.q[1] = *(const uint4*)(p + 16);
}

__device__ __forceinline__ f32x8 wmma_bf(const ABu& a, const ABu& b, f32x8 c) {
  return __builtin_amdgcn_wmma_f32_16x16x32_bf16(false, a.v, false, b.v,
                                                 (short)0, c, false, false);
}

// ---------------- weight packing: f32[rows,K] -> bf16[rows,KP] (zero padded) ----------------
__global__ void pack_bf16_kernel(const float* __restrict__ src, unsigned short* __restrict__ dst,
                                 int rows, int K, int KP)
{
  int idx = blockIdx.x * blockDim.x + threadIdx.x;
  if (idx >= rows * KP) return;
  int r = idx / KP, c = idx % KP;
  dst[idx] = (c < K) ? f2bf(src[(size_t)r * K + c]) : (unsigned short)0;
}

// ---------------- generic row GEMM: C[R,NC] = act(A[R,K] @ W[NC,K]^T + bias) ----------------
// Wp: pre-packed bf16 weights with row stride KP (compile-time). act: 0=none,1=leaky,2=relu
template<int KP>
__global__ __launch_bounds__(128) void gemm_rows_kernel(
    const float* __restrict__ A, const unsigned short* __restrict__ Wp,
    const float* __restrict__ bias, float* __restrict__ C,
    int R, int K, int NC, int act)
{
  __shared__ __align__(16) unsigned short sA[64 * KP];
  const int tid = threadIdx.x;
  const int lane = tid & 31;
  const int wave = tid >> 5;
  const int hi = lane >> 4;
  const int lm = lane & 15;
  const int blk0 = blockIdx.x * 64;

  for (int idx = tid; idx < 64 * KP; idx += 128) {
    int r = idx / KP, c = idx % KP;
    int gr = blk0 + r; if (gr >= R) gr = R - 1;
    sA[idx] = (c < K) ? f2bf(A[(size_t)gr * K + c]) : (unsigned short)0;
  }
  __syncthreads();

  constexpr int KS = KP / 32;
  const int lrow = wave * 16 + lm;
  ABu afr[KS];
#pragma unroll
  for (int ks = 0; ks < KS; ++ks)
    load_a_frag(afr[ks], sA, KP, lrow, ks * 32, hi);

  for (int nt = 0; nt * 16 < NC; ++nt) {
    f32x8 acc;
#pragma unroll
    for (int i = 0; i < 8; ++i) acc[i] = 0.f;
#pragma unroll
    for (int ks = 0; ks < KS; ++ks) {
      ABu b;
      load_b_frag(b, Wp, KP, nt * 16 + lm, ks * 32, hi);
      acc = wmma_bf(afr[ks], b, acc);
    }
#pragma unroll
    for (int r8 = 0; r8 < 8; ++r8) {
      int row = blk0 + wave * 16 + r8 + hi * 8;
      int col = nt * 16 + lm;
      if (row < R) {
        float v = acc[r8];
        if (bias) v += bias[col];
        if (act == 1) v = leakyf(v);
        else if (act == 2) v = fmaxf(v, 0.f);
        C[(size_t)row * NC + col] = v;
      }
    }
  }
}

// ---------------- fused GRU cell + ReLU: outp = relu(GRUCell(inp, hid)), H=96 ----------------
__global__ __launch_bounds__(128) void gru_relu_kernel(
    const float* __restrict__ inp, const float* __restrict__ hid,
    const unsigned short* __restrict__ wihp, const unsigned short* __restrict__ whhp,
    const float* __restrict__ bih, const float* __restrict__ bhh,
    float* __restrict__ outp, int R)
{
  __shared__ __align__(16) unsigned short sI[64 * 96];
  __shared__ __align__(16) unsigned short sH[64 * 96];
  const int tid = threadIdx.x;
  const int lane = tid & 31, wave = tid >> 5;
  const int hi = lane >> 4, lm = lane & 15;
  const int blk0 = blockIdx.x * 64;

  // vectorized bf16 staging: 24 float4 per row
  for (int idx = tid; idx < 64 * 24; idx += 128) {
    int r = idx / 24, c4 = (idx % 24) * 4;
    int gr = blk0 + r; if (gr >= R) gr = R - 1;
    pack4(&sI[r * 96 + c4], *(const float4*)&inp[(size_t)gr * 96 + c4]);
    pack4(&sH[r * 96 + c4], *(const float4*)&hid[(size_t)gr * 96 + c4]);
  }
  __syncthreads();

  const int lrow = wave * 16 + lm;
  ABu aI[3], aH[3];
#pragma unroll
  for (int ks = 0; ks < 3; ++ks) {
    load_a_frag(aI[ks], sI, 96, lrow, ks * 32, hi);
    load_a_frag(aH[ks], sH, 96, lrow, ks * 32, hi);
  }

  for (int jt = 0; jt < 6; ++jt) {
    f32x8 gi[3], gh[3];
#pragma unroll
    for (int g = 0; g < 3; ++g)
#pragma unroll
      for (int i = 0; i < 8; ++i) { gi[g][i] = 0.f; gh[g][i] = 0.f; }

#pragma unroll
    for (int ks = 0; ks < 3; ++ks) {
#pragma unroll
      for (int g = 0; g < 3; ++g) {
        ABu bI, bH;
        int col = jt * 16 + lm + g * 96;
        load_b_frag(bI, wihp, 96, col, ks * 32, hi);
        load_b_frag(bH, whhp, 96, col, ks * 32, hi);
        gi[g] = wmma_bf(aI[ks], bI, gi[g]);
        gh[g] = wmma_bf(aH[ks], bH, gh[g]);
      }
    }
#pragma unroll
    for (int r8 = 0; r8 < 8; ++r8) {
      int lr = wave * 16 + r8 + hi * 8;
      int grow = blk0 + lr;
      int col = jt * 16 + lm;
      float ir = gi[0][r8] + bih[col];
      float iz = gi[1][r8] + bih[col + 96];
      float in_ = gi[2][r8] + bih[col + 192];
      float hr = gh[0][r8] + bhh[col];
      float hz = gh[1][r8] + bhh[col + 96];
      float hn = gh[2][r8] + bhh[col + 192];
      float rr = sigm(ir + hr);
      float zz = sigm(iz + hz);
      float nn = tanhf(in_ + rr * hn);
      int hrow = grow < R ? grow : R - 1;
      float hv = hid[(size_t)hrow * 96 + col];
      float v = (1.f - zz) * nn + zz * hv;
      if (grow < R) outp[(size_t)grow * 96 + col] = fmaxf(v, 0.f);
    }
  }
}

// ---------------- fused GATEConv edge pass ----------------
// he = leaky([xh[src], e_attr] @ Wg1^T); w = exp(leaky(he.att_l + ascal[dst]))
// sEx[dst] += w; hbuf[dst] += (he @ Wg2^T) * w
__global__ __launch_bounds__(256) void gate_edge_kernel(
    const float* __restrict__ xh, const float* __restrict__ eattr,
    const int* __restrict__ src, const int* __restrict__ dst,
    const unsigned short* __restrict__ Wg1p,  // bf16 [96,128]
    const unsigned short* __restrict__ Wg2p,  // bf16 [96,96]
    const float* __restrict__ att_l, const float* __restrict__ ascal,
    float* __restrict__ sEx, float* __restrict__ hbuf, int E)
{
  __shared__ __align__(16) unsigned short sA[128 * 128];  // gathered rows, K padded to 128
  __shared__ __align__(16) unsigned short sHe[128 * 96];  // he tile
  __shared__ float sW[128];
  __shared__ int sSrc[128], sDst[128];
  const int tid = threadIdx.x;
  const int lane = tid & 31, wave = tid >> 5;
  const int hi = lane >> 4, lm = lane & 15;
  const int e0 = blockIdx.x * 128;

  if (tid < 128) {
    int e = e0 + tid; if (e >= E) e = E - 1;
    sSrc[tid] = src[e];
    sDst[tid] = dst[e];
  }
  if (tid == 0 && e0 + 128 < E) __builtin_prefetch(&src[e0 + 128], 0, 1);
  __syncthreads();

  // gathered xh rows: 24 float4 per row, vectorized convert
  for (int idx = tid; idx < 128 * 24; idx += 256) {
    int er = idx / 24, c4 = (idx % 24) * 4;
    pack4(&sA[er * 128 + c4], *(const float4*)&xh[(size_t)sSrc[er] * 96 + c4]);
  }
  // edge_attr columns 96..105 + zero pad 106..127
  for (int idx = tid; idx < 128 * 32; idx += 256) {
    int er = idx / 32, c = 96 + (idx % 32);
    int e = e0 + er; if (e >= E) e = E - 1;
    float v = (c < 106) ? eattr[(size_t)e * 10 + (c - 96)] : 0.f;
    sA[er * 128 + c] = f2bf(v);
  }
  __syncthreads();

  const int lrow = wave * 16 + lm;
  // GEMM1: he = leaky(A @ Wg1^T)
  {
    ABu af[4];
#pragma unroll
    for (int ks = 0; ks < 4; ++ks)
      load_a_frag(af[ks], sA, 128, lrow, ks * 32, hi);
#pragma unroll
    for (int nt = 0; nt < 6; ++nt) {
      f32x8 acc;
#pragma unroll
      for (int i = 0; i < 8; ++i) acc[i] = 0.f;
#pragma unroll
      for (int ks = 0; ks < 4; ++ks) {
        ABu b;
        load_b_frag(b, Wg1p, 128, nt * 16 + lm, ks * 32, hi);
        acc = wmma_bf(af[ks], b, acc);
      }
#pragma unroll
      for (int r8 = 0; r8 < 8; ++r8) {
        int row = wave * 16 + r8 + hi * 8;
        sHe[row * 96 + nt * 16 + lm] = f2bf(leakyf(acc[r8]));
      }
    }
  }
  __syncthreads();

  // attention logit + exp (single-pass softmax; max-shift omitted, activations tiny)
  if (tid < 128) {
    float dotv = 0.f;
    for (int c = 0; c < 96; ++c) dotv += bf2f(sHe[tid * 96 + c]) * att_l[c];
    int e = e0 + tid;
    int d = sDst[tid];
    float w = expf(leakyf(dotv + ascal[d]));
    if (e < E) atomicAdd(&sEx[d], w);
    sW[tid] = (e < E) ? w : 0.f;
  }
  __syncthreads();

  // GEMM2: msg = (he @ Wg2^T) * w, scatter-add into hbuf[dst]
  {
    ABu af[3];
#pragma unroll
    for (int ks = 0; ks < 3; ++ks)
      load_a_frag(af[ks], sHe, 96, lrow, ks * 32, hi);
#pragma unroll
    for (int nt = 0; nt < 6; ++nt) {
      f32x8 acc;
#pragma unroll
      for (int i = 0; i < 8; ++i) acc[i] = 0.f;
#pragma unroll
      for (int ks = 0; ks < 3; ++ks) {
        ABu b;
        load_b_frag(b, Wg2p, 96, nt * 16 + lm, ks * 32, hi);
        acc = wmma_bf(af[ks], b, acc);
      }
#pragma unroll
      for (int r8 = 0; r8 < 8; ++r8) {
        int er = wave * 16 + r8 + hi * 8;
        int e = e0 + er;
        if (e < E) {
          float v = acc[r8] * sW[er];
          atomicAdd(&hbuf[(size_t)sDst[er] * 96 + nt * 16 + lm], v);
        }
      }
    }
  }
}

// ---------------- per-row dot products with 1-2 vectors ----------------
__global__ void dots_kernel(const float* __restrict__ X, const float* __restrict__ v1,
                            const float* __restrict__ v2, float* __restrict__ o1,
                            float* __restrict__ o2, int R, int K)
{
  int r = blockIdx.x * blockDim.x + threadIdx.x;
  if (r >= R) return;
  float d1 = 0.f, d2 = 0.f;
  for (int k = 0; k < K; ++k) {
    float x = X[(size_t)r * K + k];
    d1 += x * v1[k];
    if (v2) d2 += x * v2[k];
  }
  o1[r] = d1;
  if (o2) o2[r] = d2;
}

// ---------------- atom GAT edge scatter (warp per edge) ----------------
__global__ void attn_edge_scatter(const float* __restrict__ xt, const int* __restrict__ src,
    const int* __restrict__ dst, const float* __restrict__ ps, const float* __restrict__ pd,
    float* __restrict__ sEx, float* __restrict__ acc, int E)
{
  int g = blockIdx.x * blockDim.x + threadIdx.x;
  int e = g >> 5, lane = g & 31;
  if (e >= E) return;
  int s = src[e], d = dst[e];
  float w = expf(leakyf(ps[s] + pd[d]));
  if (lane == 0) atomicAdd(&sEx[d], w);
  for (int c = lane; c < 96; c += 32)
    atomicAdd(&acc[(size_t)d * 96 + c], xt[(size_t)s * 96 + c] * w);
}

// ---------------- plain segment-sum of rows (warp per row) ----------------
__global__ void segsum_rows(const float* __restrict__ X, const int* __restrict__ seg,
                            float* __restrict__ acc, int R)
{
  int g = blockIdx.x * blockDim.x + threadIdx.x;
  int r = g >> 5, lane = g & 31;
  if (r >= R) return;
  int s = seg[r];
  for (int c = lane; c < 96; c += 32)
    atomicAdd(&acc[(size_t)s * 96 + c], X[(size_t)r * 96 + c]);
}

// ---------------- molecule attention scatter (warp per node) ----------------
__global__ void mol_attn_scatter(const float* __restrict__ xs, const int* __restrict__ batch,
    const float* __restrict__ a_src, const float* __restrict__ qd,
    float* __restrict__ sEx, float* __restrict__ acc, int R)
{
  int g = blockIdx.x * blockDim.x + threadIdx.x;
  int r = g >> 5, lane = g & 31;
  if (r >= R) return;
  int b = batch[r];
  float w = expf(leakyf(a_src[r] + qd[b]));
  if (lane == 0) atomicAdd(&sEx[b], w);
  for (int c = lane; c < 96; c += 32)
    atomicAdd(&acc[(size_t)b * 96 + c], xs[(size_t)r * 96 + c] * w);
}

// ---------------- softmax-normalize + bias + ELU ----------------
__global__ void norm_bias_elu(float* __restrict__ buf, const float* __restrict__ s,
                              const float* __restrict__ bias, int R)
{
  int idx = blockIdx.x * blockDim.x + threadIdx.x;
  if (idx >= R * 96) return;
  int r = idx / 96, c = idx % 96;
  float v = buf[idx] / (s[r] + 1e-16f) + bias[c];
  buf[idx] = eluf(v);
}

__global__ void relu_ip(float* __restrict__ b, int n)
{
  int i = blockIdx.x * blockDim.x + threadIdx.x;
  if (i < n) b[i] = fmaxf(b[i], 0.f);
}

// ---------------- temperature encoder ----------------
__global__ void temp_encode(const float* __restrict__ temp, const float* __restrict__ w1,
    const float* __restrict__ b1, const float* __restrict__ w2, const float* __restrict__ b2,
    float* __restrict__ te, int G)
{
  int idx = blockIdx.x * blockDim.x + threadIdx.x;
  if (idx >= G * 64) return;
  int g = idx / 64, j = idx % 64;
  float tn = (temp[g] - 25.f) / 35.f;
  float acc = b2[j];
  for (int k = 0; k < 64; ++k)
    acc += fmaxf(tn * w1[k] + b1[k], 0.f) * w2[j * 64 + k];
  te[idx] = acc;
}

// ---------------- fusion (layernorm + linear + relu) + MLP head, block per graph ----------------
__global__ __launch_bounds__(256) void head_kernel(
    const float* __restrict__ graphb, const float* __restrict__ te,
    const float* __restrict__ ln_g, const float* __restrict__ ln_b,
    const float* __restrict__ fw, const float* __restrict__ fb,
    const float* __restrict__ w1, const float* __restrict__ b1,
    const float* __restrict__ w2, const float* __restrict__ b2,
    const float* __restrict__ w3, const float* __restrict__ b3,
    float* __restrict__ outp)
{
  __shared__ float comb[256], red[256], fused[192], hh1[64], hh2[64];
  int g = blockIdx.x, tid = threadIdx.x;
  comb[tid] = (tid < 192) ? graphb[(size_t)g * 192 + tid] : te[(size_t)g * 64 + (tid - 192)];
  __syncthreads();
  red[tid] = comb[tid]; __syncthreads();
  for (int s = 128; s > 0; s >>= 1) { if (tid < s) red[tid] += red[tid + s]; __syncthreads(); }
  float mu = red[0] / 256.f; __syncthreads();
  float dd = comb[tid] - mu;
  red[tid] = dd * dd; __syncthreads();
  for (int s = 128; s > 0; s >>= 1) { if (tid < s) red[tid] += red[tid + s]; __syncthreads(); }
  float var = red[0] / 256.f; __syncthreads();
  comb[tid] = dd * rsqrtf(var + 1e-5f) * ln_g[tid] + ln_b[tid];
  __syncthreads();
  if (tid < 192) {
    float a = fb[tid];
    for (int k = 0; k < 256; ++k) a += comb[k] * fw[tid * 256 + k];
    fused[tid] = fmaxf(a, 0.f);
  }
  __syncthreads();
  if (tid < 64) {
    float a = b1[tid];
    for (int k = 0; k < 192; ++k) a += fused[k] * w1[tid * 192 + k];
    hh1[tid] = fmaxf(a, 0.f);
  }
  __syncthreads();
  if (tid < 64) {
    float a = b2[tid];
    for (int k = 0; k < 64; ++k) a += hh1[k] * w2[tid * 64 + k];
    hh2[tid] = fmaxf(a, 0.f);
  }
  __syncthreads();
  if (tid < 6) {
    float a = b3[tid];
    for (int k = 0; k < 64; ++k) a += hh2[k] * w3[tid * 64 + k];
    outp[(size_t)g * 6 + tid] = a;
  }
}

// ---------------- host driver ----------------
extern "C" void kernel_launch(void* const* d_in, const int* in_sizes, int n_in,
                              void* d_out, int out_size, void* d_ws, size_t ws_size,
                              hipStream_t stream)
{
  (void)in_sizes; (void)n_in; (void)out_size; (void)ws_size;

  // inputs (setup_inputs insertion order; params flattened recursively in source order)
  const float* x      = (const float*)d_in[0];
  const int*   ei     = (const int*)d_in[1];
  const float* eattr  = (const float*)d_in[2];
  const int*   batch  = (const int*)d_in[3];
  const float* temp   = (const float*)d_in[4];
  const float* lin1_w = (const float*)d_in[5];
  const float* lin1_b = (const float*)d_in[6];
  const float* g_l1   = (const float*)d_in[7];   // gate.lin1_w [96,106]
  const float* g_l2   = (const float*)d_in[8];   // gate.lin2_w [96,96]
  const float* g_attl = (const float*)d_in[9];
  const float* g_attr = (const float*)d_in[10];
  const float* g_bias = (const float*)d_in[11];
  const float* g0_wih = (const float*)d_in[12];
  const float* g0_whh = (const float*)d_in[13];
  const float* g0_bih = (const float*)d_in[14];
  const float* g0_bhh = (const float*)d_in[15];
  const float* at_w   = (const float*)d_in[16];
  const float* at_as  = (const float*)d_in[17];
  const float* at_ad  = (const float*)d_in[18];
  const float* at_b   = (const float*)d_in[19];
  const float* g1_wih = (const float*)d_in[20];
  const float* g1_whh = (const float*)d_in[21];
  const float* g1_bih = (const float*)d_in[22];
  const float* g1_bhh = (const float*)d_in[23];
  const float* m_w    = (const float*)d_in[24];
  const float* m_as   = (const float*)d_in[25];
  const float* m_ad   = (const float*)d_in[26];
  const float* m_b    = (const float*)d_in[27];
  const float* mg_wih = (const float*)d_in[28];
  const float* mg_whh = (const float*)d_in[29];
  const float* mg_bih = (const float*)d_in[30];
  const float* mg_bhh = (const float*)d_in[31];
  const float* lin2_w = (const float*)d_in[32];
  const float* lin2_b = (const float*)d_in[33];
  const float* t1w    = (const float*)d_in[34];
  const float* t1b    = (const float*)d_in[35];
  const float* t2w    = (const float*)d_in[36];
  const float* t2b    = (const float*)d_in[37];
  const float* lng    = (const float*)d_in[38];
  const float* lnb    = (const float*)d_in[39];
  const float* fw     = (const float*)d_in[40];
  const float* fbv    = (const float*)d_in[41];
  const float* h1w    = (const float*)d_in[42];
  const float* h1b    = (const float*)d_in[43];
  const float* h2w    = (const float*)d_in[44];
  const float* h2b    = (const float*)d_in[45];
  const float* h3w    = (const float*)d_in[46];
  const float* h3b    = (const float*)d_in[47];

  const int* src = ei;
  const int* dst = ei + E_EDGES;

  // workspace layout: f32 region (~160 MB) then bf16 packed-weight region (~0.5 MB)
  float* ws = (float*)d_ws;
  float* xh   = ws; ws += (size_t)N_NODES * H_F;
  float* hbuf = ws; ws += (size_t)N_NODES * H_F;
  float* xt   = ws; ws += (size_t)N_NODES * H_F;   // also xs in mol phase
  float* sEx  = ws; ws += N_NODES;                 // softmax denominators (N >= G)
  float* d1   = ws; ws += N_NODES;                 // att_r dots / ps / a_src
  float* d2   = ws; ws += N_NODES;                 // pd
  float* outg = ws; ws += (size_t)G_GRAPHS * H_F;
  float* hg   = ws; ws += (size_t)G_GRAPHS * H_F;
  float* grb  = ws; ws += (size_t)G_GRAPHS * OUT_F;
  float* te   = ws; ws += (size_t)G_GRAPHS * TED_F;
  float* qd   = ws; ws += G_GRAPHS;

  unsigned short* wp = (unsigned short*)ws;
  unsigned short* lin1p  = wp; wp += 96 * 64;
  unsigned short* g1p    = wp; wp += 96 * 128;
  unsigned short* g2p    = wp; wp += 96 * 96;
  unsigned short* g0wihp = wp; wp += 288 * 96;
  unsigned short* g0whhp = wp; wp += 288 * 96;
  unsigned short* g1wihp = wp; wp += 288 * 96;
  unsigned short* g1whhp = wp; wp += 288 * 96;
  unsigned short* mgwihp = wp; wp += 288 * 96;
  unsigned short* mgwhhp = wp; wp += 288 * 96;
  unsigned short* atwp   = wp; wp += 96 * 96;
  unsigned short* mwp    = wp; wp += 96 * 96;
  unsigned short* lin2p  = wp; wp += 192 * 96;

  dim3 b128(128), b256(256);
  auto packN = [](int n) { return dim3((unsigned)((n + 255) / 256)); };

  // 0) pack all GEMM weights to bf16 (padded K)
  pack_bf16_kernel<<<packN(96 * 64),   b256, 0, stream>>>(lin1_w, lin1p, 96, IN_F, 64);
  pack_bf16_kernel<<<packN(96 * 128),  b256, 0, stream>>>(g_l1, g1p, 96, 106, 128);
  pack_bf16_kernel<<<packN(96 * 96),   b256, 0, stream>>>(g_l2, g2p, 96, 96, 96);
  pack_bf16_kernel<<<packN(288 * 96),  b256, 0, stream>>>(g0_wih, g0wihp, 288, 96, 96);
  pack_bf16_kernel<<<packN(288 * 96),  b256, 0, stream>>>(g0_whh, g0whhp, 288, 96, 96);
  pack_bf16_kernel<<<packN(288 * 96),  b256, 0, stream>>>(g1_wih, g1wihp, 288, 96, 96);
  pack_bf16_kernel<<<packN(288 * 96),  b256, 0, stream>>>(g1_whh, g1whhp, 288, 96, 96);
  pack_bf16_kernel<<<packN(288 * 96),  b256, 0, stream>>>(mg_wih, mgwihp, 288, 96, 96);
  pack_bf16_kernel<<<packN(288 * 96),  b256, 0, stream>>>(mg_whh, mgwhhp, 288, 96, 96);
  pack_bf16_kernel<<<packN(96 * 96),   b256, 0, stream>>>(at_w, atwp, 96, 96, 96);
  pack_bf16_kernel<<<packN(96 * 96),   b256, 0, stream>>>(m_w, mwp, 96, 96, 96);
  pack_bf16_kernel<<<packN(192 * 96),  b256, 0, stream>>>(lin2_w, lin2p, 192, 96, 96);

  // 1) xh = leaky(x @ lin1_w^T + b)
  gemm_rows_kernel<64><<<dim3((N_NODES + 63) / 64), b128, 0, stream>>>(
      x, lin1p, lin1_b, xh, N_NODES, IN_F, H_F, 1);

  // 2) GATEConv
  dots_kernel<<<packN(N_NODES), b256, 0, stream>>>(
      xh, g_attr, nullptr, d1, nullptr, N_NODES, H_F);
  hipMemsetAsync(sEx, 0, N_NODES * sizeof(float), stream);
  hipMemsetAsync(hbuf, 0, (size_t)N_NODES * H_F * sizeof(float), stream);
  gate_edge_kernel<<<dim3(E_EDGES / 128), b256, 0, stream>>>(
      xh, eattr, src, dst, g1p, g2p, g_attl, d1, sEx, hbuf, E_EDGES);
  norm_bias_elu<<<packN(N_NODES * H_F), b256, 0, stream>>>(hbuf, sEx, g_bias, N_NODES);
  gru_relu_kernel<<<dim3((N_NODES + 63) / 64), b128, 0, stream>>>(
      hbuf, xh, g0wihp, g0whhp, g0_bih, g0_bhh, xh, N_NODES);

  // 3) atom GATConv
  gemm_rows_kernel<96><<<dim3((N_NODES + 63) / 64), b128, 0, stream>>>(
      xh, atwp, nullptr, xt, N_NODES, H_F, H_F, 0);
  dots_kernel<<<packN(N_NODES), b256, 0, stream>>>(
      xt, at_as, at_ad, d1, d2, N_NODES, H_F);
  hipMemsetAsync(sEx, 0, N_NODES * sizeof(float), stream);
  hipMemsetAsync(hbuf, 0, (size_t)N_NODES * H_F * sizeof(float), stream);
  attn_edge_scatter<<<dim3((E_EDGES * 32) / 256), b256, 0, stream>>>(
      xt, src, dst, d1, d2, sEx, hbuf, E_EDGES);
  norm_bias_elu<<<packN(N_NODES * H_F), b256, 0, stream>>>(hbuf, sEx, at_b, N_NODES);
  gru_relu_kernel<<<dim3((N_NODES + 63) / 64), b128, 0, stream>>>(
      hbuf, xh, g1wihp, g1whhp, g1_bih, g1_bhh, xh, N_NODES);

  // 4) molecule readout
  hipMemsetAsync(outg, 0, (size_t)G_GRAPHS * H_F * sizeof(float), stream);
  segsum_rows<<<dim3((N_NODES * 32) / 256), b256, 0, stream>>>(xh, batch, outg, N_NODES);
  relu_ip<<<packN(G_GRAPHS * H_F), b256, 0, stream>>>(outg, G_GRAPHS * H_F);
  gemm_rows_kernel<96><<<dim3((N_NODES + 63) / 64), b128, 0, stream>>>(
      xh, mwp, nullptr, xt, N_NODES, H_F, H_F, 0);               // xs
  dots_kernel<<<packN(N_NODES), b256, 0, stream>>>(
      xt, m_as, nullptr, d1, nullptr, N_NODES, H_F);             // a_src

  for (int t = 0; t < TSTEPS; ++t) {
    gemm_rows_kernel<96><<<dim3((G_GRAPHS + 63) / 64), b128, 0, stream>>>(
        outg, mwp, nullptr, hg, G_GRAPHS, H_F, H_F, 0);          // out @ w^T
    dots_kernel<<<packN(G_GRAPHS), b256, 0, stream>>>(
        hg, m_ad, nullptr, qd, nullptr, G_GRAPHS, H_F);          // qd
    hipMemsetAsync(sEx, 0, G_GRAPHS * sizeof(float), stream);
    hipMemsetAsync(hg, 0, (size_t)G_GRAPHS * H_F * sizeof(float), stream);
    mol_attn_scatter<<<dim3((N_NODES * 32) / 256), b256, 0, stream>>>(
        xt, batch, d1, qd, sEx, hg, N_NODES);
    norm_bias_elu<<<packN(G_GRAPHS * H_F), b256, 0, stream>>>(hg, sEx, m_b, G_GRAPHS);
    gru_relu_kernel<<<dim3((G_GRAPHS + 63) / 64), b128, 0, stream>>>(
        hg, outg, mgwihp, mgwhhp, mg_bih, mg_bhh, outg, G_GRAPHS);
  }

  // 5) graph projection + temperature + fusion + head
  gemm_rows_kernel<96><<<dim3((G_GRAPHS + 63) / 64), b128, 0, stream>>>(
      outg, lin2p, lin2_b, grb, G_GRAPHS, H_F, OUT_F, 0);
  temp_encode<<<packN(G_GRAPHS * 64), b256, 0, stream>>>(
      temp, t1w, t1b, t2w, t2b, te, G_GRAPHS);
  head_kernel<<<dim3(G_GRAPHS), b256, 0, stream>>>(
      grb, te, lng, lnb, fw, fbv, h1w, h1b, h2w, h2b, h3w, h3b, (float*)d_out);
}
